// NeRF_19456201851144
// MI455X (gfx1250) — matrix operand order repaired and checked by reading, not compiled, not run on hardware
//
#include <hip/hip_runtime.h>
#include <math.h>

// ---------------------------------------------------------------------------
// Types for WMMA fragments (wave32, gfx1250)
// ---------------------------------------------------------------------------
typedef __attribute__((ext_vector_type(16))) _Float16 v16h;
typedef __attribute__((ext_vector_type(8)))  _Float16 v8h;
typedef __attribute__((ext_vector_type(8)))  float    v8f;

union Frag16 {
    v16h v;
    struct { v8h lo, hi; } p;
};

#define LDK   168          // LDS row stride (f16 elems); 336B/row -> conflict-friendly
#define MTILE 64           // points per workgroup
#define MLP_BLOCK 256      // 8 waves

__device__ __forceinline__ v8f wmma_step(const Frag16& a, const Frag16& b, v8f acc)
{
    return __builtin_amdgcn_wmma_f32_16x16x32_f16(
        /*neg_a=*/false, a.v, /*neg_b=*/false, b.v,
        /*c_mod=*/(short)0, acc, /*reuse_a=*/false, /*reuse_b=*/false);
}

// ---------------------------------------------------------------------------
// Single 16x16 tile (used for the small Ws / Wr heads)
// A layout (16-bit 16x32): lane m = lane&15; lanes<16: K{0..7,16..23}, lanes>=16: K{8..15,24..31}
// B layout (16-bit 32x16): lane n = lane&15; lanes<16: K{0..7,16..23}, lanes>=16: K{8..15,24..31}
// ---------------------------------------------------------------------------
__device__ __forceinline__ v8f mma_tile(const _Float16* X, const _Float16* wrow,
                                        int K, int mtile, int lane)
{
    v8f acc = {0.f, 0.f, 0.f, 0.f, 0.f, 0.f, 0.f, 0.f};
    const int akb = (lane >> 4) << 3;
    const int bk  = (lane >> 4) << 3;
    const _Float16* arow = X + (mtile * 16 + (lane & 15)) * LDK;
#pragma unroll
    for (int k0 = 0; k0 < K; k0 += 32) {
        Frag16 a, b;
        a.p.lo = *(const v8h*)(arow + k0 + akb);
        a.p.hi = *(const v8h*)(arow + k0 + akb + 16);
        b.p.lo = *(const v8h*)(wrow + k0 + bk);
        b.p.hi = *(const v8h*)(wrow + k0 + bk + 16);
        acc = wmma_step(a, b, acc);
    }
    return acc;
}

// C/D layout: VGPR r -> M = r + 8*(lane>=16), N = lane&15
__device__ __forceinline__ void store_tile_lds(_Float16* Y, v8f acc, float bv,
                                               int ntile, int mtile, int lane, bool do_relu)
{
    const int n = lane & 15;
    const int mbase = mtile * 16 + ((lane >> 4) << 3);
#pragma unroll
    for (int r = 0; r < 8; ++r) {
        float v = acc[r] + bv;
        if (do_relu) v = v > 0.f ? v : 0.f;
        Y[(mbase + r) * LDK + ntile * 16 + n] = (_Float16)v;
    }
}

// ---------------------------------------------------------------------------
// Dense layer, one N-tile per call: preload all B fragments into registers,
// sweep M-tiles of A from LDS against them (4x weight-fetch reuse).
// ---------------------------------------------------------------------------
template <int KSTEPS>
__device__ __forceinline__ void dense_ntile(const _Float16* __restrict__ X,
                                            _Float16* __restrict__ Y,
                                            const _Float16* __restrict__ wrow, // + lane's row
                                            float bv, int ntile,
                                            int mstart, int mcount,
                                            int lane, bool do_relu)
{
    const int akb = (lane >> 4) << 3;
    const int bk  = (lane >> 4) << 3;
    Frag16 b[KSTEPS];
#pragma unroll
    for (int k = 0; k < KSTEPS; ++k) {
        b[k].p.lo = *(const v8h*)(wrow + k * 32 + bk);
        b[k].p.hi = *(const v8h*)(wrow + k * 32 + bk + 16);
    }
    for (int mt = mstart; mt < mstart + mcount; ++mt) {
        const _Float16* arow = X + (mt * 16 + (lane & 15)) * LDK;
        v8f acc = {0.f, 0.f, 0.f, 0.f, 0.f, 0.f, 0.f, 0.f};
#pragma unroll
        for (int k = 0; k < KSTEPS; ++k) {
            Frag16 a;
            a.p.lo = *(const v8h*)(arow + k * 32 + akb);
            a.p.hi = *(const v8h*)(arow + k * 32 + akb + 16);
            acc = wmma_step(a, b[k], acc);
        }
        store_tile_lds(Y, acc, bv, ntile, mt, lane, do_relu);
    }
}

// ---------------------------------------------------------------------------
// Weight transpose + pad:  src fp32 [K][N] row-major  ->  dst f16 [Np][Kp]
// ---------------------------------------------------------------------------
__global__ void k_transpose_pad(const float* __restrict__ src, _Float16* __restrict__ dst,
                                int K, int N, int Kp, int Np)
{
    int idx = blockIdx.x * blockDim.x + threadIdx.x;
    if (idx >= Kp * Np) return;
    int n = idx / Kp;
    int k = idx - n * Kp;
    float v = (k < K && n < N) ? src[k * N + n] : 0.f;
    dst[n * Kp + k] = (_Float16)v;
}

// ---------------------------------------------------------------------------
// Per-ray prep: |d|, view-dir embedding (27 -> padded 32, f16)
// ---------------------------------------------------------------------------
__global__ void k_prep(const float* __restrict__ rays_d, _Float16* __restrict__ ve16,
                       float* __restrict__ dnorm, int nrays)
{
    int r = blockIdx.x * blockDim.x + threadIdx.x;
    if (r >= nrays) return;
    float dx = rays_d[r * 3 + 0], dy = rays_d[r * 3 + 1], dz = rays_d[r * 3 + 2];
    float nrm = sqrtf(dx * dx + dy * dy + dz * dz);
    dnorm[r] = nrm;
    float inv = 1.f / nrm;
    float v[3] = {dx * inv, dy * inv, dz * inv};
    _Float16* o = ve16 + r * 32;
    o[0] = (_Float16)v[0]; o[1] = (_Float16)v[1]; o[2] = (_Float16)v[2];
#pragma unroll
    for (int l = 0; l < 4; ++l) {
        float f = (float)(1 << l);
#pragma unroll
        for (int j = 0; j < 3; ++j) {
            float a = v[j] * f;
            o[3 + 6 * l + j]     = (_Float16)__sinf(a);
            o[3 + 6 * l + 3 + j] = (_Float16)__cosf(a);
        }
    }
#pragma unroll
    for (int c = 27; c < 32; ++c) o[c] = (_Float16)0.f;
}

// ---------------------------------------------------------------------------
// Fused positional-embed + full NeRF MLP. One workgroup = 64 points, 8 waves.
// Weight blob (f16, transposed+padded) offsets (elems):
//   W0T 0 [128][64] | W1T 8192 | W2T 24576 | W3T 40960 [128][128]
//   WfT 57344 [128][128] | WvT 73728 [64][160] | WsT 83968 [16][128] | WrT 86016 [16][64]
// ---------------------------------------------------------------------------
__global__ void __launch_bounds__(MLP_BLOCK)
k_mlp(const float* __restrict__ rays_o, const float* __restrict__ rays_d,
      const float* __restrict__ zbuf, int nsamp,
      const _Float16* __restrict__ WT,
      const float* __restrict__ bW0, const float* __restrict__ bW1,
      const float* __restrict__ bW2, const float* __restrict__ bW3,
      const float* __restrict__ bWf, const float* __restrict__ bWs,
      const float* __restrict__ bWv, const float* __restrict__ bWr,
      const _Float16* __restrict__ ve16,
      float* __restrict__ sigma_out, float* __restrict__ rgb_out)
{
    __shared__ _Float16 sA[MTILE * LDK];
    __shared__ _Float16 sB[MTILE * LDK];

    const int tid  = threadIdx.x;
    const int lane = tid & 31;
    const int wv   = tid >> 5;
    const int p0   = blockIdx.x * MTILE;
    const int lrow = lane & 15;

    const _Float16* W0T = WT;
    const _Float16* W1T = WT + 8192;
    const _Float16* W2T = WT + 24576;
    const _Float16* W3T = WT + 40960;
    const _Float16* WfT = WT + 57344;
    const _Float16* WvT = WT + 73728;
    const _Float16* WsT = WT + 83968;
    const _Float16* WrT = WT + 86016;

    // ---- positional embedding of 64 points into sA cols [0,64) ----
    for (int idx = tid; idx < MTILE * 64; idx += MLP_BLOCK) {
        int m = idx >> 6, c = idx & 63;
        int p = p0 + m;
        int r = p / nsamp;
        int s = p - r * nsamp;
        float z = zbuf ? zbuf[p] : (float)s * (1.f / 63.f);
        float val = 0.f;
        if (c < 63) {
            int d, l = 0; bool is_cos = false;
            if (c < 3) { d = c; }
            else {
                int e = c - 3; l = e / 6; int j = e - l * 6;
                is_cos = j >= 3; d = is_cos ? j - 3 : j;
            }
            float pos = rays_o[r * 3 + d] + rays_d[r * 3 + d] * z;
            if (c < 3) val = pos;
            else {
                float a = pos * (float)(1 << l);
                val = is_cos ? __cosf(a) : __sinf(a);
            }
        }
        sA[m * LDK + c] = (_Float16)val;
    }
    __syncthreads();

    // ---- 128-wide layers: wave w owns N-tile w, sweeps 4 M-tiles ----
    // pe(64) -> h0
    dense_ntile<2>(sA, sB, W0T + (wv * 16 + lrow) * 64,  bW0[wv * 16 + lrow], wv, 0, 4, lane, true);
    __syncthreads();
    dense_ntile<4>(sB, sA, W1T + (wv * 16 + lrow) * 128, bW1[wv * 16 + lrow], wv, 0, 4, lane, true);
    __syncthreads();
    dense_ntile<4>(sA, sB, W2T + (wv * 16 + lrow) * 128, bW2[wv * 16 + lrow], wv, 0, 4, lane, true);
    __syncthreads();
    dense_ntile<4>(sB, sA, W3T + (wv * 16 + lrow) * 128, bW3[wv * 16 + lrow], wv, 0, 4, lane, true);
    __syncthreads();
    // h3 now in sA.

    // ---- feat = h3 @ Wf + bf (no relu) into sB[:,0:128) ----
    dense_ntile<4>(sA, sB, WfT + (wv * 16 + lrow) * 128, bWf[wv * 16 + lrow], wv, 0, 4, lane, false);

    // ---- append ve into sB[:,128:160) ----
    for (int idx = tid; idx < MTILE * 32; idx += MLP_BLOCK) {
        int m = idx >> 5, c = idx & 31;
        int r = (p0 + m) / nsamp;
        sB[m * LDK + 128 + c] = ve16[r * 32 + c];
    }

    // ---- sigma = h3 @ Ws + bs (padded 16-col tile, col 0 valid): waves 0..3 ----
    if (wv < 4) {
        const _Float16* wrow = WsT + lrow * 128;
        v8f acc = mma_tile(sA, wrow, 128, wv, lane);
        if (lrow == 0) {
            float bsv = bWs[0];
            int mbase = wv * 16 + ((lane >> 4) << 3);
#pragma unroll
            for (int rr = 0; rr < 8; ++rr)
                sigma_out[p0 + mbase + rr] = acc[rr] + bsv;
        }
    }
    __syncthreads();

    // ---- hv = relu([feat,ve](160) @ Wv + bv) -> sA[:,0:64) ----
    // 4 N-tiles x 4 M-tiles: wave w -> ntile w&3, M-tiles [(w>>2)*2, +2)
    {
        int ntile  = wv & 3;
        int mstart = (wv >> 2) * 2;
        dense_ntile<5>(sB, sA, WvT + (ntile * 16 + lrow) * 160, bWv[ntile * 16 + lrow],
                       ntile, mstart, 2, lane, true);
    }
    __syncthreads();

    // ---- rgb = hv @ Wr + br (padded 16-col tile, cols 0..2 valid): waves 0..3 ----
    if (wv < 4) {
        const _Float16* wrow = WrT + lrow * 64;
        v8f acc = mma_tile(sA, wrow, 64, wv, lane);
        if (lrow < 3) {
            float brv = bWr[lrow];
            int mbase = wv * 16 + ((lane >> 4) << 3);
#pragma unroll
            for (int rr = 0; rr < 8; ++rr)
                rgb_out[(p0 + mbase + rr) * 3 + lrow] = acc[rr] + brv;
        }
    }
}

// ---------------------------------------------------------------------------
// Volume rendering: one thread per ray (sequential cumprod)
// ---------------------------------------------------------------------------
__global__ void k_render(const float* __restrict__ sigma, const float* __restrict__ rgbraw,
                         const float* __restrict__ zbuf, int nsamp,
                         const float* __restrict__ dnorm,
                         float* __restrict__ out_rgb, float* __restrict__ wout, int nrays)
{
    int r = blockIdx.x * blockDim.x + threadIdx.x;
    if (r >= nrays) return;
    float T = 1.f, cr = 0.f, cg = 0.f, cb = 0.f;
    float dn = dnorm[r];
    for (int s = 0; s < nsamp; ++s) {
        int p = r * nsamp + s;
        float z = zbuf ? zbuf[p] : (float)s * (1.f / 63.f);
        float dist;
        if (s + 1 < nsamp) {
            float zn = zbuf ? zbuf[p + 1] : (float)(s + 1) * (1.f / 63.f);
            dist = (zn - z) * dn;
        } else {
            dist = 1e10f * dn;
        }
        float sg = sigma[p]; sg = sg > 0.f ? sg : 0.f;
        float a  = 1.f - __expf(-sg * dist);
        float w  = a * T;
        T *= (1.f - a + 1e-10f);
        float rr = rgbraw[p * 3 + 0], gg = rgbraw[p * 3 + 1], bb = rgbraw[p * 3 + 2];
        cr += w * (1.f / (1.f + __expf(-rr)));
        cg += w * (1.f / (1.f + __expf(-gg)));
        cb += w * (1.f / (1.f + __expf(-bb)));
        if (wout) wout[p] = w;
    }
    out_rgb[r * 3 + 0] = cr;
    out_rgb[r * 3 + 1] = cg;
    out_rgb[r * 3 + 2] = cb;
}

// ---------------------------------------------------------------------------
// Inverse-CDF importance sampling + sorted merge -> z_all[r][192]
// ---------------------------------------------------------------------------
__global__ void k_samplepdf(const float* __restrict__ wbuf, float* __restrict__ zall, int nrays)
{
    int r = blockIdx.x * blockDim.x + threadIdx.x;
    if (r >= nrays) return;

    float wloc[62];
    float cdf[63];
    float sum = 0.f;
    for (int i = 0; i < 62; ++i) {
        float w = wbuf[r * 64 + 1 + i] + 1e-5f;
        wloc[i] = w; sum += w;
    }
    float inv = 1.f / sum;
    cdf[0] = 0.f;
    for (int i = 0; i < 62; ++i) cdf[i + 1] = cdf[i] + wloc[i] * inv;

    float zs[128];
    for (int i = 0; i < 128; ++i) {
        float u = (float)i * (1.f / 127.f);
        int lo = 0, hi = 63;                 // searchsorted right over cdf[0..62]
        while (lo < hi) {
            int mid = (lo + hi) >> 1;
            if (cdf[mid] > u) hi = mid; else lo = mid + 1;
        }
        int inds  = lo;
        int below = inds - 1; if (below < 0) below = 0;
        int above = inds;     if (above > 62) above = 62;
        float c0 = cdf[below], c1 = cdf[above];
        float b0 = ((float)below + 0.5f) * (1.f / 63.f);
        float b1 = ((float)above + 0.5f) * (1.f / 63.f);
        float den = c1 - c0; if (den < 1e-5f) den = 1.f;
        float t = (u - c0) / den;
        zs[i] = b0 + t * (b1 - b0);
    }

    float* za = zall + r * 192;
    int i = 0, j = 0;
    for (int k = 0; k < 192; ++k) {
        float a = (i < 64)  ? (float)i * (1.f / 63.f) : 3.4e38f;
        float b = (j < 128) ? zs[j]                   : 3.4e38f;
        if (a <= b) { za[k] = a; ++i; } else { za[k] = b; ++j; }
    }
}

// ---------------------------------------------------------------------------
// Host launcher
// Input order: rays_o, rays_d, then per net (coarse, fine):
//   W0,b0,W1,b1,W2,b2,W3,b3,Ws,bs,Wf,bf,Wv,bv,Wr,br
// Output: rgb_coarse (nrays*3) || rgb_fine (nrays*3)
// ---------------------------------------------------------------------------
extern "C" void kernel_launch(void* const* d_in, const int* in_sizes, int n_in,
                              void* d_out, int out_size, void* d_ws, size_t ws_size,
                              hipStream_t stream)
{
    (void)n_in; (void)out_size; (void)ws_size;
    const float* rays_o = (const float*)d_in[0];
    const float* rays_d = (const float*)d_in[1];
    const int nrays = in_sizes[0] / 3;          // 4096
    const int NS_C = 64, NS_F = 192;

    // ---- workspace carve-up ----
    char* ws = (char*)d_ws;
    size_t off = 0;
    const size_t WT_ELEMS = 87040;
    _Float16* WTc  = (_Float16*)(ws + off); off += WT_ELEMS * 2;
    _Float16* WTf  = (_Float16*)(ws + off); off += WT_ELEMS * 2;
    _Float16* ve16 = (_Float16*)(ws + off); off += (size_t)nrays * 32 * 2;
    off = (off + 255) & ~(size_t)255;
    float* dnorm   = (float*)(ws + off); off += (size_t)nrays * 4;
    float* zall    = (float*)(ws + off); off += (size_t)nrays * NS_F * 4;
    float* wcoarse = (float*)(ws + off); off += (size_t)nrays * NS_C * 4;
    float* sig     = (float*)(ws + off); off += (size_t)nrays * NS_F * 4;
    float* rgbraw  = (float*)(ws + off); off += (size_t)nrays * NS_F * 3 * 4;

    // ---- convert & transpose weights (both nets) ----
    auto tp = [&](const void* src, _Float16* dst, int K, int N, int Kp, int Np) {
        int tot = Kp * Np;
        k_transpose_pad<<<(tot + 255) / 256, 256, 0, stream>>>(
            (const float*)src, dst, K, N, Kp, Np);
    };
    for (int net = 0; net < 2; ++net) {
        int b = 2 + net * 16;
        _Float16* WT = net ? WTf : WTc;
        tp(d_in[b + 0],  WT + 0,     63, 128,  64, 128);   // W0
        tp(d_in[b + 2],  WT + 8192, 128, 128, 128, 128);   // W1
        tp(d_in[b + 4],  WT + 24576,128, 128, 128, 128);   // W2
        tp(d_in[b + 6],  WT + 40960,128, 128, 128, 128);   // W3
        tp(d_in[b + 10], WT + 57344,128, 128, 128, 128);   // Wf
        tp(d_in[b + 12], WT + 73728,155,  64, 160,  64);   // Wv
        tp(d_in[b + 8],  WT + 83968,128,   1, 128,  16);   // Ws (padded cols)
        tp(d_in[b + 14], WT + 86016, 64,   3,  64,  16);   // Wr (padded cols)
    }

    // ---- per-ray prep ----
    k_prep<<<(nrays + 255) / 256, 256, 0, stream>>>(rays_d, ve16, dnorm, nrays);

    // ---- coarse pass ----
    {
        int b = 2;
        k_mlp<<<nrays * NS_C / MTILE, MLP_BLOCK, 0, stream>>>(
            rays_o, rays_d, (const float*)nullptr, NS_C, WTc,
            (const float*)d_in[b + 1], (const float*)d_in[b + 3],
            (const float*)d_in[b + 5], (const float*)d_in[b + 7],
            (const float*)d_in[b + 11], (const float*)d_in[b + 9],
            (const float*)d_in[b + 13], (const float*)d_in[b + 15],
            ve16, sig, rgbraw);
        k_render<<<(nrays + 255) / 256, 256, 0, stream>>>(
            sig, rgbraw, (const float*)nullptr, NS_C, dnorm,
            (float*)d_out, wcoarse, nrays);
        k_samplepdf<<<(nrays + 255) / 256, 256, 0, stream>>>(wcoarse, zall, nrays);
    }

    // ---- fine pass ----
    {
        int b = 2 + 16;
        k_mlp<<<nrays * NS_F / MTILE, MLP_BLOCK, 0, stream>>>(
            rays_o, rays_d, zall, NS_F, WTf,
            (const float*)d_in[b + 1], (const float*)d_in[b + 3],
            (const float*)d_in[b + 5], (const float*)d_in[b + 7],
            (const float*)d_in[b + 11], (const float*)d_in[b + 9],
            (const float*)d_in[b + 13], (const float*)d_in[b + 15],
            ve16, sig, rgbraw);
        k_render<<<(nrays + 255) / 256, 256, 0, stream>>>(
            sig, rgbraw, zall, NS_F, dnorm,
            (float*)d_out + (size_t)nrays * 3, (float*)nullptr, nrays);
    }
}
